// VectorQuantizer_4226247819579
// MI455X (gfx1250) — matrix operand order repaired and checked by reading, not compiled, not run on hardware
//
#include <hip/hip_runtime.h>
#include <hip/hip_bf16.h>

typedef __attribute__((ext_vector_type(16))) _Float16 v16h;
typedef __attribute__((ext_vector_type(8)))  _Float16 v8h;
typedef __attribute__((ext_vector_type(8)))  float    v8f;

#define C_DIM   256
#define K_CODES 1024
#define HW      1024      // 32*32
#define N_PIX   32768     // 32*32*32
#define M_TILE  32        // pixels per wave (2 WMMA M-tiles)
#define ZQ_ELEMS ((size_t)N_PIX * C_DIM)   // 8388608 floats, then indices

// ---------------------------------------------------------------------------
// Kernel 1: L2-normalize codebook rows. One wave (32 lanes) per row.
// Emits f32 normalized rows (for final gather) and f16 rows (WMMA B operand).
// ---------------------------------------------------------------------------
__global__ void vq_norm_weight(const float* __restrict__ w,
                               float* __restrict__ wn32,
                               _Float16* __restrict__ wn16) {
    const int row  = blockIdx.x * (blockDim.x >> 5) + (threadIdx.x >> 5);
    const int lane = threadIdx.x & 31;
    const float* src = w + (size_t)row * C_DIM + lane * 8;

    float4 a = *(const float4*)(src);
    float4 b = *(const float4*)(src + 4);
    float ss = a.x*a.x + a.y*a.y + a.z*a.z + a.w*a.w
             + b.x*b.x + b.y*b.y + b.z*b.z + b.w*b.w;
#pragma unroll
    for (int m = 16; m >= 1; m >>= 1) ss += __shfl_xor(ss, m, 32);
    const float scale = 1.0f / fmaxf(sqrtf(ss), 1e-12f);

    float4 an = make_float4(a.x*scale, a.y*scale, a.z*scale, a.w*scale);
    float4 bn = make_float4(b.x*scale, b.y*scale, b.z*scale, b.w*scale);
    float* dst32 = wn32 + (size_t)row * C_DIM + lane * 8;
    *(float4*)(dst32)     = an;
    *(float4*)(dst32 + 4) = bn;

    v8h h;
    h[0] = (_Float16)an.x; h[1] = (_Float16)an.y;
    h[2] = (_Float16)an.z; h[3] = (_Float16)an.w;
    h[4] = (_Float16)bn.x; h[5] = (_Float16)bn.y;
    h[6] = (_Float16)bn.z; h[7] = (_Float16)bn.w;
    *(v8h*)(wn16 + (size_t)row * C_DIM + lane * 8) = h;
}

// ---------------------------------------------------------------------------
// Kernel 2: one wave handles 32 pixels (two 16-row M-tiles sharing each
// B fragment -> 2 WMMAs per global_load_b128, halving L2 codebook traffic).
// ---------------------------------------------------------------------------
__global__ void __launch_bounds__(32)
vq_main(const float* __restrict__ z_e,
        const float* __restrict__ wn32,
        const _Float16* __restrict__ wn16,
        float* __restrict__ out) {
    __shared__ float lds_a[M_TILE][C_DIM + 4];   // +4 pad: conflict-free [p][c]
    __shared__ int   lds_idx[M_TILE];

    const int lane = threadIdx.x & 31;
    const int p    = lane & 15;   // row-in-tile (phase1/2) / code column (phase3)
    const int hi   = lane >> 4;

    const int npix0 = blockIdx.x * M_TILE;   // 32 pixels per wave
    const int b     = npix0 >> 10;           // HW == 1024, tiles never straddle b
    const int hw0   = npix0 & (HW - 1);

    // ---- Phase 1: load 32x256 pixel block (coalesced), per-pixel sum-sq ----
    const size_t base = (size_t)b * C_DIM * HW + hw0 + p;
    float scale[2];
#pragma unroll
    for (int t = 0; t < 2; ++t) {
        float ss = 0.0f;
#pragma unroll 4
        for (int c0 = 0; c0 < C_DIM; c0 += 2) {
            const int c = c0 + hi;
            const float v = z_e[base + 16 * t + (size_t)c * HW];
            lds_a[16 * t + p][c] = v;
            ss += v * v;
        }
        ss += __shfl_xor(ss, 16, 32);             // combine channel parities
        scale[t] = 1.0f / fmaxf(sqrtf(ss), 1e-12f);
    }
    __syncthreads();

    // ---- Phase 2: build A fragments (16x32 A, contiguous k order as in B) ----
    // lane group hi: halves 0..15 = K k0+16*hi+j
    v16h afrag[2][8];
#pragma unroll
    for (int t = 0; t < 2; ++t) {
        const float s = scale[t];
        const float* arow = &lds_a[16 * t + p][0];
#pragma unroll
        for (int kt = 0; kt < 8; ++kt) {
            const int base_c = kt * 32 + hi * 16;
            const float4 x0 = *(const float4*)(arow + base_c);
            const float4 x1 = *(const float4*)(arow + base_c + 4);
            const float4 x2 = *(const float4*)(arow + base_c + 8);
            const float4 x3 = *(const float4*)(arow + base_c + 12);
            v16h f;
            f[0]  = (_Float16)(x0.x * s); f[1]  = (_Float16)(x0.y * s);
            f[2]  = (_Float16)(x0.z * s); f[3]  = (_Float16)(x0.w * s);
            f[4]  = (_Float16)(x1.x * s); f[5]  = (_Float16)(x1.y * s);
            f[6]  = (_Float16)(x1.z * s); f[7]  = (_Float16)(x1.w * s);
            f[8]  = (_Float16)(x2.x * s); f[9]  = (_Float16)(x2.y * s);
            f[10] = (_Float16)(x2.z * s); f[11] = (_Float16)(x2.w * s);
            f[12] = (_Float16)(x3.x * s); f[13] = (_Float16)(x3.y * s);
            f[14] = (_Float16)(x3.z * s); f[15] = (_Float16)(x3.w * s);
            afrag[t][kt] = f;
        }
    }

    // ---- Phase 3: 64 code tiles; each B fragment feeds BOTH M-tiles ----
    float bestv[2][8];
    int   besti[2][8];
#pragma unroll
    for (int t = 0; t < 2; ++t)
#pragma unroll
        for (int i = 0; i < 8; ++i) { bestv[t][i] = -3.402823e38f; besti[t][i] = 0; }

    for (int jt = 0; jt < K_CODES / 16; ++jt) {
        const int code_base = jt << 4;
        // B fragment: column n = p, 16 contiguous K halves selected by hi.
        const _Float16* brow = wn16 + (size_t)(code_base + p) * C_DIM + hi * 16;
        v8f acc0 = {0.f, 0.f, 0.f, 0.f, 0.f, 0.f, 0.f, 0.f};
        v8f acc1 = {0.f, 0.f, 0.f, 0.f, 0.f, 0.f, 0.f, 0.f};
#pragma unroll
        for (int kt = 0; kt < 8; ++kt) {
            const v16h bfrag = *(const v16h*)(brow + kt * 32);   // 32B aligned, L2-hot
            acc0 = __builtin_amdgcn_wmma_f32_16x16x32_f16(
                       false, afrag[0][kt], false, bfrag, (short)0, acc0, false, false);
            acc1 = __builtin_amdgcn_wmma_f32_16x16x32_f16(
                       false, afrag[1][kt], false, bfrag, (short)0, acc1, false, false);
        }
        // C/D layout: VGPR i -> pixel row (i + 8*hi), column = code_base + p
#pragma unroll
        for (int i = 0; i < 8; ++i) {
            const float v0 = acc0[i];
            if (v0 > bestv[0][i]) { bestv[0][i] = v0; besti[0][i] = code_base + p; }
            const float v1 = acc1[i];
            if (v1 > bestv[1][i]) { bestv[1][i] = v1; besti[1][i] = code_base + p; }
        }
    }

    // ---- Phase 4: reduce across the 16 columns within each half-wave ----
#pragma unroll
    for (int m = 8; m >= 1; m >>= 1) {
#pragma unroll
        for (int t = 0; t < 2; ++t)
#pragma unroll
            for (int i = 0; i < 8; ++i) {
                const float ov = __shfl_xor(bestv[t][i], m, 32);
                const int   oi = __shfl_xor(besti[t][i], m, 32);
                if (ov > bestv[t][i] || (ov == bestv[t][i] && oi < besti[t][i])) {
                    bestv[t][i] = ov; besti[t][i] = oi;
                }
            }
    }
    if (p == 0) {
#pragma unroll
        for (int t = 0; t < 2; ++t)
#pragma unroll
            for (int i = 0; i < 8; ++i)
                lds_idx[16 * t + hi * 8 + i] = besti[t][i];
    }
    __syncthreads();

    // ---- Phase 5: outputs ----
    out[ZQ_ELEMS + npix0 + lane] = (float)lds_idx[lane];   // indices (tuple tail)

    const int code0 = lds_idx[p];
    const int code1 = lds_idx[16 + p];
    const float* wrow0 = wn32 + (size_t)code0 * C_DIM;
    const float* wrow1 = wn32 + (size_t)code1 * C_DIM;
#pragma unroll 4
    for (int c0 = 0; c0 < C_DIM; c0 += 2) {
        const int c = c0 + hi;
        const size_t o = base + (size_t)c * HW;
        out[o]      = wrow0[c];
        out[o + 16] = wrow1[c];
    }
}

// ---------------------------------------------------------------------------
extern "C" void kernel_launch(void* const* d_in, const int* in_sizes, int n_in,
                              void* d_out, int out_size, void* d_ws, size_t ws_size,
                              hipStream_t stream) {
    const float* z_e = (const float*)d_in[0];       // [32,256,32,32] f32
    const float* w   = (const float*)d_in[1];       // [1024,256] f32
    float* out = (float*)d_out;                     // z_q (8388608) ++ indices (32768)

    float*    wn32 = (float*)d_ws;                                           // 1 MB
    _Float16* wn16 = (_Float16*)((char*)d_ws + (size_t)K_CODES * C_DIM * 4); // +512 KB

    // Kernel 1: 1024 rows, 8 waves per 256-thread block
    vq_norm_weight<<<K_CODES / 8, 256, 0, stream>>>(w, wn32, wn16);
    // Kernel 2: one wave per 32-pixel tile
    vq_main<<<N_PIX / M_TILE, 32, 0, stream>>>(z_e, wn32, wn16, out);
}
